// Pointer_6743098655483
// MI455X (gfx1250) — matrix operand order repaired and checked by reading, not compiled, not run on hardware
//
#include <hip/hip_runtime.h>
#include <cstddef>

#define HH 512
#define NN 1024
#define BB 128

typedef __attribute__((ext_vector_type(16))) __bf16 v16bf;
typedef __attribute__((ext_vector_type(8)))  float  v8f;

union BF { v16bf v; uint4 q[2]; };

#define WMMA_BF(A, B, C) \
    __builtin_amdgcn_wmma_f32_16x16x32_bf16(false, (A), false, (B), (short)0, (C), false, false)

__device__ __forceinline__ float waveRed(float v) {
    #pragma unroll
    for (int o = 16; o > 0; o >>= 1) v += __shfl_down(v, o, 32);
    return v;
}

__device__ __forceinline__ unsigned short f2bf(float x) {  // RNE float->bf16
    unsigned u = __float_as_uint(x);
    u += 0x7FFFu + ((u >> 16) & 1u);
    return (unsigned short)(u >> 16);
}
__device__ __forceinline__ float bf2f(unsigned short b) {
    return __uint_as_float(((unsigned)b) << 16);
}

// ---------------- kernel 1: t2[b,h] = sum_k attn_W[h, H+k] * mc[b,k] (fp32 exact) --------
__global__ __launch_bounds__(256) void k_t2(const float* __restrict__ mc,
                                            const float* __restrict__ W,
                                            float* __restrict__ t2) {
    __shared__ float mcl[HH];
    const int tid = threadIdx.x, b = blockIdx.x;
    for (int i = tid; i < HH; i += 256) mcl[i] = mc[b * HH + i];
    __syncthreads();
    const int wave = tid >> 5, lane = tid & 31;
    for (int h = wave; h < HH; h += 8) {
        const float* wr = W + (size_t)h * (2 * HH) + HH;  // W2 half
        float p = 0.f;
        for (int i = lane; i < HH; i += 32) p += wr[i] * mcl[i];
        p = waveRed(p);
        if (lane == 0) t2[b * HH + h] = p;
    }
}

// ---------------- kernel 2 (bf16x3 WMMA): scores[b,n] ----------------
// scores[b,n] = sum_h v[h]*tanh((W1@sn[b])[h,n] + t2[b,h])
// W1@sn emulated as Ah*Bh + Ah*Bl + Al*Bh with hi/lo bf16 split, fp32 accumulate.
// Block: 128 threads (4 waves). Strip: 64 n-columns, full K in LDS.
// Wave (hp,np) owns a 32h x 32n register tile (2x2 WMMA tiles) per h-pass of 64 rows.
#define SN_ST 72    // bf16 elements per sn LDS row   (64 + pad, 16B-aligned rows)
#define WL_ST 520   // bf16 elements per W  LDS row   (512 + pad, 16B-aligned rows)
#define SCORES_SMEM (2*(HH*SN_ST*2) + 2*(64*WL_ST*2) + (2*HH)*4 + 2*64*4)

__global__ __launch_bounds__(128, 1) void k_scores(const float* __restrict__ sn,
                                                   const float* __restrict__ W,
                                                   const float* __restrict__ av,
                                                   const float* __restrict__ t2,
                                                   float* __restrict__ scores) {
    extern __shared__ unsigned short smem_us[];
    unsigned short* snH = smem_us;                 // [512][SN_ST] bf16 hi
    unsigned short* snL = snH + HH * SN_ST;        // [512][SN_ST] bf16 lo
    unsigned short* wH  = snL + HH * SN_ST;        // [64][WL_ST]  bf16 hi
    unsigned short* wL  = wH + 64 * WL_ST;         // [64][WL_ST]  bf16 lo
    float* t2l       = (float*)(wL + 64 * WL_ST);  // [512]
    float* vl        = t2l + HH;                   // [512]
    float* scorePart = vl + HH;                    // [2][64]

    const int tid  = threadIdx.x;
    const int wave = tid >> 5, lane = tid & 31;
    const int b = blockIdx.y, n0 = blockIdx.x * 64;

    // ---- stage sn strip (fp32 -> hi/lo bf16), 16 threads/row, float4 each ----
    const float* snb = sn + (size_t)b * ((size_t)HH * NN) + n0;
    {
        const int c4 = (tid & 15) * 4;
        for (int r = tid >> 4; r < HH; r += 8) {
            float4 s = *(const float4*)(snb + (size_t)r * NN + c4);
            unsigned short hx = f2bf(s.x), hy = f2bf(s.y), hz = f2bf(s.z), hw = f2bf(s.w);
            uint2 ph; ph.x = (unsigned)hx | ((unsigned)hy << 16);
                      ph.y = (unsigned)hz | ((unsigned)hw << 16);
            unsigned short lx = f2bf(s.x - bf2f(hx)), ly = f2bf(s.y - bf2f(hy));
            unsigned short lz = f2bf(s.z - bf2f(hz)), lw = f2bf(s.w - bf2f(hw));
            uint2 pl; pl.x = (unsigned)lx | ((unsigned)ly << 16);
                      pl.y = (unsigned)lz | ((unsigned)lw << 16);
            *(uint2*)(snH + r * SN_ST + c4) = ph;
            *(uint2*)(snL + r * SN_ST + c4) = pl;
        }
    }
    for (int i = tid; i < HH; i += 128) { t2l[i] = t2[b * HH + i]; vl[i] = av[i]; }
    __syncthreads();

    const int mA = lane & 15;     // A row in tile / D col in tile
    const int hl = lane >> 4;     // half-wave
    const int hp = wave >> 1;     // h-tile-pair select (0/1)
    const int np = wave & 1;      // n-subtile-pair select (0/1)
    const int kA = hl * 8;        // A-frag K offset (ISA 16-bit A layout)
    const int rowB = mA + hl * 16;   // unused; B-frag uses same K order as A (K along elements)
    const int nb0 = np * 32, nb1 = nb0 + 16;

    const unsigned short* aH0 = wH + (hp * 32 + mA) * WL_ST;
    const unsigned short* aH1 = aH0 + 16 * WL_ST;
    const unsigned short* aL0 = wL + (hp * 32 + mA) * WL_ST;
    const unsigned short* aL1 = aL0 + 16 * WL_ST;

    float sAcc0 = 0.f, sAcc1 = 0.f;

    for (int pass = 0; pass < HH / 64; ++pass) {
        const int h0 = pass * 64;
        // ---- stage W1 panel rows h0..h0+63 -> hi/lo bf16 (2 threads/row) ----
        {
            const int row = tid >> 1;
            const int halfsel = (tid & 1) * 256;
            const float* wr = W + (size_t)(h0 + row) * (2 * HH) + halfsel;
            unsigned short* dH = wH + row * WL_ST + halfsel;
            unsigned short* dL = wL + row * WL_ST + halfsel;
            #pragma unroll 4
            for (int j = 0; j < 64; ++j) {
                float4 s = *(const float4*)(wr + j * 4);
                unsigned short hx = f2bf(s.x), hy = f2bf(s.y), hz = f2bf(s.z), hw = f2bf(s.w);
                uint2 ph; ph.x = (unsigned)hx | ((unsigned)hy << 16);
                          ph.y = (unsigned)hz | ((unsigned)hw << 16);
                unsigned short lx = f2bf(s.x - bf2f(hx)), ly = f2bf(s.y - bf2f(hy));
                unsigned short lz = f2bf(s.z - bf2f(hz)), lw = f2bf(s.w - bf2f(hw));
                uint2 pl; pl.x = (unsigned)lx | ((unsigned)ly << 16);
                          pl.y = (unsigned)lz | ((unsigned)lw << 16);
                *(uint2*)(dH + j * 4) = ph;
                *(uint2*)(dL + j * 4) = pl;
            }
        }
        __syncthreads();

        v8f accA00 = {}, accA01 = {}, accA10 = {}, accA11 = {};
        v8f accB00 = {}, accB01 = {}, accB10 = {}, accB11 = {};

        for (int k0 = 0; k0 < HH; k0 += 32) {
            const int ka = k0 + kA;
            BF a0h, a0l, a1h, a1l;
            a0h.q[0] = *(const uint4*)(aH0 + ka); a0h.q[1] = *(const uint4*)(aH0 + ka + 16);
            a1h.q[0] = *(const uint4*)(aH1 + ka); a1h.q[1] = *(const uint4*)(aH1 + ka + 16);
            a0l.q[0] = *(const uint4*)(aL0 + ka); a0l.q[1] = *(const uint4*)(aL0 + ka + 16);
            a1l.q[0] = *(const uint4*)(aL1 + ka); a1l.q[1] = *(const uint4*)(aL1 + ka + 16);

            const unsigned short* sH = snH + (size_t)(k0 + kA) * SN_ST + mA;
            const unsigned short* sL = snL + (size_t)(k0 + kA) * SN_ST + mA;
            auto gat = [&](const unsigned short* p) {
                BF r;
                #pragma unroll
                for (int qq = 0; qq < 2; ++qq) {
                    #pragma unroll
                    for (int w = 0; w < 4; ++w) {
                        const int kk = qq * 16 + 2 * w;
                        ((unsigned*)&r.q[qq])[w] = (unsigned)p[kk * SN_ST] | ((unsigned)p[(kk + 1) * SN_ST] << 16);
                    }
                }
                return r;
            };
            BF b0h = gat(sH + nb0), b1h = gat(sH + nb1);
            BF b0l = gat(sL + nb0), b1l = gat(sL + nb1);

            // main (hi*hi) terms
            accA00 = WMMA_BF(a0h.v, b0h.v, accA00);
            accA01 = WMMA_BF(a0h.v, b1h.v, accA01);
            accA10 = WMMA_BF(a1h.v, b0h.v, accA10);
            accA11 = WMMA_BF(a1h.v, b1h.v, accA11);
            // correction terms (hi*lo + lo*hi)
            accB00 = WMMA_BF(a0h.v, b0l.v, accB00); accB00 = WMMA_BF(a0l.v, b0h.v, accB00);
            accB01 = WMMA_BF(a0h.v, b1l.v, accB01); accB01 = WMMA_BF(a0l.v, b1h.v, accB01);
            accB10 = WMMA_BF(a1h.v, b0l.v, accB10); accB10 = WMMA_BF(a1l.v, b0h.v, accB10);
            accB11 = WMMA_BF(a1h.v, b1l.v, accB11); accB11 = WMMA_BF(a1l.v, b1h.v, accB11);
        }

        // fold tiles into per-column score partials (D vgpr i -> M = i + 8*hl)
        const int hbase = h0 + hp * 32 + hl * 8;
        #pragma unroll
        for (int i = 0; i < 8; ++i) {
            const int hg0 = hbase + i;
            const int hg1 = hbase + 16 + i;
            sAcc0 += vl[hg0] * tanhf(accA00[i] + accB00[i] + t2l[hg0]);
            sAcc1 += vl[hg0] * tanhf(accA01[i] + accB01[i] + t2l[hg0]);
            sAcc0 += vl[hg1] * tanhf(accA10[i] + accB10[i] + t2l[hg1]);
            sAcc1 += vl[hg1] * tanhf(accA11[i] + accB11[i] + t2l[hg1]);
        }
        __syncthreads();
    }

    // combine half-waves (rows 0-7 vs 8-15 of each tile live in lane, lane+16)
    const float o0 = __shfl_down(sAcc0, 16, 32);
    const float o1 = __shfl_down(sAcc1, 16, 32);
    if (lane < 16) {
        scorePart[hp * 64 + nb0 + mA]      = sAcc0 + o0;
        scorePart[hp * 64 + nb1 + mA]      = sAcc1 + o1;
    }
    __syncthreads();
    // combine the two h-tile-pair waves (fixed order -> deterministic)
    if (tid < 64) scores[b * NN + n0 + tid] = scorePart[tid] + scorePart[64 + tid];
}

// ---------------- kernel 3: softmax over n ----------------
__global__ __launch_bounds__(256) void k_softmax(const float* __restrict__ scores,
                                                 float* __restrict__ attns) {
    __shared__ float red[256];
    const int tid = threadIdx.x, b = blockIdx.x;
    float x[4];
    #pragma unroll
    for (int j = 0; j < 4; ++j) x[j] = scores[b * NN + tid + j * 256];
    float m = fmaxf(fmaxf(x[0], x[1]), fmaxf(x[2], x[3]));
    red[tid] = m; __syncthreads();
    for (int s = 128; s > 0; s >>= 1) {
        if (tid < s) red[tid] = fmaxf(red[tid], red[tid + s]);
        __syncthreads();
    }
    const float gm = red[0];
    __syncthreads();
    float e[4], ss = 0.f;
    #pragma unroll
    for (int j = 0; j < 4; ++j) { e[j] = expf(x[j] - gm); ss += e[j]; }
    red[tid] = ss; __syncthreads();
    for (int s = 128; s > 0; s >>= 1) {
        if (tid < s) red[tid] += red[tid + s];
        __syncthreads();
    }
    const float inv = 1.f / red[0];
    #pragma unroll
    for (int j = 0; j < 4; ++j) attns[b * NN + tid + j * 256] = e[j] * inv;
}

// ---------------- kernel 4: context[b,h] = sum_n attns[b,n]*sn[b,h,n] ----------------
__global__ __launch_bounds__(512) void k_context(const float* __restrict__ sn,
                                                 const float* __restrict__ attns,
                                                 float* __restrict__ ctx) {
    __shared__ float al[NN];
    const int tid = threadIdx.x, b = blockIdx.x;
    for (int i = tid; i < NN; i += 512) al[i] = attns[b * NN + i];
    __syncthreads();
    const int wave = tid >> 5, lane = tid & 31;
    for (int h = wave; h < HH; h += 16) {
        const float* row = sn + (size_t)b * ((size_t)HH * NN) + (size_t)h * NN;
        float p = 0.f;
        #pragma unroll
        for (int i = 0; i < 8; ++i) {
            const int idx = i * 128 + lane * 4;
            float4 s = *(const float4*)(row + idx);
            p += s.x * al[idx] + s.y * al[idx + 1] + s.z * al[idx + 2] + s.w * al[idx + 3];
        }
        p = waveRed(p);
        if (lane == 0) ctx[b * HH + h] = p;
    }
}

// ---------------- kernel 5: two FC + ReLU layers ----------------
__global__ __launch_bounds__(512) void k_fc(const float* __restrict__ mc,
                                            const float* __restrict__ ctx,
                                            const float* __restrict__ W1,
                                            const float* __restrict__ b1,
                                            const float* __restrict__ W2,
                                            const float* __restrict__ b2,
                                            float* __restrict__ out2) {
    __shared__ float xl[2 * HH];
    __shared__ float o1l[HH];
    const int tid = threadIdx.x, b = blockIdx.x;
    for (int i = tid; i < HH; i += 512) { xl[i] = mc[b * HH + i]; xl[HH + i] = ctx[b * HH + i]; }
    __syncthreads();
    const int wave = tid >> 5, lane = tid & 31;
    for (int h = wave; h < HH; h += 16) {
        const float* wr = W1 + (size_t)h * (2 * HH);
        float p = 0.f;
        for (int i = lane; i < 2 * HH; i += 32) p += wr[i] * xl[i];
        p = waveRed(p);
        if (lane == 0) o1l[h] = fmaxf(p + b1[h], 0.f);
    }
    __syncthreads();
    for (int h = wave; h < HH; h += 16) {
        const float* wr = W2 + (size_t)h * HH;
        float p = 0.f;
        for (int i = lane; i < HH; i += 32) p += wr[i] * o1l[i];
        p = waveRed(p);
        if (lane == 0) out2[b * HH + h] = fmaxf(p + b2[h], 0.f);
    }
}

// ---------------- kernel 6: probs[b,n] = sum_h ptr_v[h]*tanh(sn[b,h,n] + out2[b,h]) ------
__global__ __launch_bounds__(256) void k_probs(const float* __restrict__ sn,
                                               const float* __restrict__ pv,
                                               const float* __restrict__ out2,
                                               float* __restrict__ probs) {
    __shared__ float o2l[HH];
    __shared__ float pvl[HH];
    const int tid = threadIdx.x, b = blockIdx.y;
    for (int i = tid; i < HH; i += 256) { o2l[i] = out2[b * HH + i]; pvl[i] = pv[i]; }
    __syncthreads();
    const int n = blockIdx.x * 256 + tid;
    const float* base = sn + (size_t)b * ((size_t)HH * NN) + n;
    float acc = 0.f;
    #pragma unroll 4
    for (int h = 0; h < HH; ++h)
        acc += pvl[h] * tanhf(base[(size_t)h * NN] + o2l[h]);
    probs[b * NN + n] = acc;
}

extern "C" void kernel_launch(void* const* d_in, const int* in_sizes, int n_in,
                              void* d_out, int out_size, void* d_ws, size_t ws_size,
                              hipStream_t stream) {
    const float* mc   = (const float*)d_in[0];  // (B,H)
    const float* sn   = (const float*)d_in[1];  // (B,H,N)
    const float* aW   = (const float*)d_in[2];  // (H,2H)
    const float* av   = (const float*)d_in[3];  // (H,)
    const float* pv   = (const float*)d_in[4];  // (H,)
    const float* f1W  = (const float*)d_in[5];  // (H,2H)
    const float* f1b  = (const float*)d_in[6];  // (H,)
    const float* f2W  = (const float*)d_in[7];  // (H,H)
    const float* f2b  = (const float*)d_in[8];  // (H,)
    float* out = (float*)d_out;                 // (B,N)

    float* ws     = (float*)d_ws;
    float* t2     = ws;                         // B*H
    float* scores = t2 + BB * HH;               // B*N
    float* attns  = scores + BB * NN;           // B*N
    float* ctx    = attns + BB * NN;            // B*H
    float* o2     = ctx + BB * HH;              // B*H

    k_t2<<<BB, 256, 0, stream>>>(mc, aW, t2);
    k_scores<<<dim3(NN / 64, BB), 128, SCORES_SMEM, stream>>>(sn, aW, av, t2, scores);
    k_softmax<<<BB, 256, 0, stream>>>(scores, attns);
    k_context<<<BB, 512, 0, stream>>>(sn, attns, ctx);
    k_fc<<<BB, 512, 0, stream>>>(mc, ctx, f1W, f1b, f2W, f2b, o2);
    k_probs<<<dim3(NN / 256, BB), 256, 0, stream>>>(sn, pv, o2, out);
}
